// GRU_80874234183964
// MI455X (gfx1250) — matrix operand order repaired and checked by reading, not compiled, not run on hardware
//
#include <hip/hip_runtime.h>
#include <hip/hip_bf16.h>

typedef __attribute__((ext_vector_type(16))) __bf16 v16bf;
typedef __attribute__((ext_vector_type(8)))  float  v8f;

#define SEQ  4096
#define IND  2048
#define HID  2048
#define G3   (3*HID)   /* 6144 */
#define OUTN 2048
#define NBLK 48        /* persistent scan blocks: 48 blk * 8 waves * 16 cols = 6144 */

union FragU { uint4 q[2]; v16bf v; };

__device__ __forceinline__ unsigned short f2bf(float f) {
    unsigned u = __float_as_uint(f);
    unsigned r = u + 0x7FFFu + ((u >> 16) & 1u);   // round-to-nearest-even
    return (unsigned short)(r >> 16);
}

// ---------------------------------------------------------------- init / cvt
__global__ void init_counter(unsigned* cnt) { *cnt = 0u; }

__global__ void f32_to_bf16(const float* __restrict__ src,
                            unsigned short* __restrict__ dst, size_t n) {
    size_t i = (size_t)blockIdx.x * blockDim.x + threadIdx.x;
    size_t stride = (size_t)gridDim.x * blockDim.x;
    for (; i < n; i += stride) dst[i] = f2bf(src[i]);
}

// -------------------------------------------------- Phase 1: gi = x @ w_ih^T + b_ih
// grid (48, 32), 256 threads = 8 waves; each wave: 16 rows x 128 cols of C.
__global__ __launch_bounds__(256)
void gi_gemm(const unsigned short* __restrict__ xb,
             const unsigned short* __restrict__ wihb,
             const float* __restrict__ b_ih,
             float* __restrict__ gi) {
    const int lane = threadIdx.x & 31;
    const int wid  = threadIdx.x >> 5;
    const int l16  = lane & 15;
    const int half = lane >> 4;
    const int m0 = blockIdx.y * 128 + wid * 16;
    const int n0 = blockIdx.x * 128;

    v8f acc[8] = {};

    const uint4* aptr = (const uint4*)(xb + (size_t)(m0 + l16) * IND);
    for (int k0 = 0; k0 < IND; k0 += 32) {
        const int kb = k0 + half * 8;          // A: lanes 0-15 K=[0,8)+[16,24); 16-31 K=[8,16)+[24,32)
        FragU a;
        a.q[0] = aptr[kb >> 3];
        a.q[1] = aptr[(kb >> 3) + 2];
        const int kc = k0 + half * 8;          // B: lanes 0-15 K=[0,8)+[16,24); 16-31 K=[8,16)+[24,32)
#pragma unroll
        for (int j = 0; j < 8; ++j) {
            const int n = n0 + j * 16 + l16;
            const uint4* bptr = (const uint4*)(wihb + (size_t)n * IND);
            FragU b;
            b.q[0] = bptr[kc >> 3];
            b.q[1] = bptr[(kc >> 3) + 2];
            acc[j] = __builtin_amdgcn_wmma_f32_16x16x32_bf16(
                false, a.v, false, b.v, (short)0, acc[j], false, false);
        }
    }
#pragma unroll
    for (int j = 0; j < 8; ++j) {
        const int col = n0 + j * 16 + l16;
        const float bias = b_ih[col];
#pragma unroll
        for (int r = 0; r < 8; ++r) {
            const int row = m0 + r + half * 8;     // C layout: VGPR r = rows r / r+8
            gi[(size_t)row * G3 + col] = acc[j][r] + bias;
        }
    }
}

// -------------------------------------------------- Phase 2: persistent GRU scan
__global__ __launch_bounds__(256)
void gru_scan(const unsigned short* __restrict__ whhb,
              const float* __restrict__ gi,
              const float* __restrict__ h0,
              const float* __restrict__ b_hh,
              float* __restrict__ ghbuf,      // 2 * G3 (double buffered)
              float* __restrict__ hlast,
              unsigned* __restrict__ cnt) {
    __shared__ float          h32[HID];
    __shared__ unsigned short hbf[HID];

    const int tid  = threadIdx.x;
    const int lane = tid & 31;
    const int wid  = tid >> 5;
    const int l16  = lane & 15;
    const int half = lane >> 4;
    const int n0   = (blockIdx.x * 8 + wid) * 16;        // this wave's 16 output columns
    const unsigned keep = (l16 == 0) ? 0xFFFFFFFFu : 0u; // only A row M=0 is real

    for (int j = tid; j < HID; j += 256) {
        float v = h0[j];
        h32[j] = v;
        hbf[j] = f2bf(v);
    }
    __syncthreads();

    const uint4* bptr = (const uint4*)(whhb + (size_t)(n0 + l16) * HID);
    const uint4* hptr = (const uint4*)hbf;
    const float  bhh  = (lane < 16) ? b_hh[n0 + lane] : 0.f;

    for (int t = 0; t < SEQ; ++t) {
        // ---- gh tile: M=1 GEMV on the matrix pipe, every weight read once ----
        v8f acc = {};
        for (int k0 = 0; k0 < HID; k0 += 32) {
            const int kb = k0 + half * 8;
            FragU a;
            a.q[0] = hptr[kb >> 3];
            a.q[1] = hptr[(kb >> 3) + 2];
            a.q[0].x &= keep; a.q[0].y &= keep; a.q[0].z &= keep; a.q[0].w &= keep;
            a.q[1].x &= keep; a.q[1].y &= keep; a.q[1].z &= keep; a.q[1].w &= keep;
            const int kc = k0 + half * 8;
            FragU b;
            b.q[0] = bptr[kc >> 3];
            b.q[1] = bptr[(kc >> 3) + 2];
            acc = __builtin_amdgcn_wmma_f32_16x16x32_bf16(
                false, a.v, false, b.v, (short)0, acc, false, false);
        }
        float* gh = ghbuf + (size_t)(t & 1) * G3;
        if (lane < 16) gh[n0 + lane] = acc[0] + bhh;     // C row 0 = lanes 0-15, VGPR 0

        // ---- one device-wide barrier per step (monotonic counter) ----
        __syncthreads();
        if (tid == 0) {
            __threadfence();
            __hip_atomic_fetch_add(cnt, 1u, __ATOMIC_RELEASE, __HIP_MEMORY_SCOPE_AGENT);
            const unsigned target = (unsigned)(t + 1) * NBLK;
            while (__hip_atomic_load(cnt, __ATOMIC_ACQUIRE, __HIP_MEMORY_SCOPE_AGENT) < target)
                __builtin_amdgcn_s_sleep(1);
        }
        __syncthreads();

        // ---- gate math: every block redundantly updates its own LDS h ----
        const float* git = gi + (size_t)t * G3;
        for (int j = tid; j < HID; j += 256) {
            const float pr = git[j]           + gh[j];
            const float pz = git[j + HID]     + gh[j + HID];
            const float r  = 1.f / (1.f + __expf(-pr));
            const float z  = 1.f / (1.f + __expf(-pz));
            const float n  = tanhf(git[j + 2 * HID] + r * gh[j + 2 * HID]);
            const float h  = h32[j];
            const float hn = (1.f - z) * n + z * h;
            h32[j] = hn;
            hbf[j] = f2bf(hn);
        }
        __syncthreads();
    }

    if (blockIdx.x == 0)
        for (int j = tid; j < HID; j += 256) hlast[j] = h32[j];
}

// -------------------------------------------------- Phase 3: tiny MLP GEMVs
__global__ __launch_bounds__(256)
void fc_row(const float* __restrict__ in, const float* __restrict__ W,
            const float* __restrict__ bias, float* __restrict__ out,
            int K, int do_relu) {
    __shared__ float red[256];
    const int i = blockIdx.x;
    const float* w = W + (size_t)i * K;
    float s = 0.f;
    for (int k = threadIdx.x; k < K; k += 256) s += in[k] * w[k];
    red[threadIdx.x] = s;
    __syncthreads();
    for (int off = 128; off > 0; off >>= 1) {
        if (threadIdx.x < off) red[threadIdx.x] += red[threadIdx.x + off];
        __syncthreads();
    }
    if (threadIdx.x == 0) {
        float v = red[0] + (bias ? bias[i] : 0.f);
        if (do_relu) v = fmaxf(v, 0.f);
        out[i] = v;
    }
}

// -------------------------------------------------------------------- launch
extern "C" void kernel_launch(void* const* d_in, const int* in_sizes, int n_in,
                              void* d_out, int out_size, void* d_ws, size_t ws_size,
                              hipStream_t stream) {
    const float* x     = (const float*)d_in[0];
    const float* h0    = (const float*)d_in[1];
    const float* w_ih  = (const float*)d_in[2];
    const float* w_hh  = (const float*)d_in[3];
    const float* b_ih  = (const float*)d_in[4];
    const float* b_hh  = (const float*)d_in[5];
    const float* fc_w1 = (const float*)d_in[6];
    const float* fc_b1 = (const float*)d_in[7];
    const float* fc_w2 = (const float*)d_in[8];
    float* out = (float*)d_out;

    char* ws = (char*)d_ws;
    size_t off = 0;
    float*          gi    = (float*)(ws + off);          off += (size_t)SEQ * G3 * 4;   // 100.7 MB
    unsigned short* xb    = (unsigned short*)(ws + off); off += (size_t)SEQ * IND * 2;  //  16.8 MB
    unsigned short* wihb  = (unsigned short*)(ws + off); off += (size_t)G3  * IND * 2;  //  25.2 MB
    unsigned short* whhb  = (unsigned short*)(ws + off); off += (size_t)G3  * HID * 2;  //  25.2 MB
    float*          ghbuf = (float*)(ws + off);          off += (size_t)2 * G3 * 4;
    float*          hlast = (float*)(ws + off);          off += (size_t)HID * 4;
    float*          h1    = (float*)(ws + off);          off += (size_t)HID * 4;
    unsigned*       cnt   = (unsigned*)(ws + off);       off += 64;

    init_counter<<<1, 1, 0, stream>>>(cnt);
    f32_to_bf16<<<1024, 256, 0, stream>>>(x,    xb,   (size_t)SEQ * IND);
    f32_to_bf16<<<1024, 256, 0, stream>>>(w_ih, wihb, (size_t)G3  * IND);
    f32_to_bf16<<<1024, 256, 0, stream>>>(w_hh, whhb, (size_t)G3  * HID);

    gi_gemm<<<dim3(G3 / 128, SEQ / 128), 256, 0, stream>>>(xb, wihb, b_ih, gi);

    gru_scan<<<NBLK, 256, 0, stream>>>(whhb, gi, h0, b_hh, ghbuf, hlast, cnt);

    fc_row<<<HID,  256, 0, stream>>>(hlast, fc_w1, fc_b1,  h1,  HID, 1);
    fc_row<<<OUTN, 256, 0, stream>>>(h1,    fc_w2, nullptr, out, HID, 0);
}
